// Multihead_Attention_60765197304477
// MI455X (gfx1250) — matrix operand order repaired and checked by reading, not compiled, not run on hardware
//
#include <hip/hip_runtime.h>

typedef __attribute__((ext_vector_type(16))) _Float16 v16h;
typedef __attribute__((ext_vector_type(8)))  _Float16 v8h;
typedef __attribute__((ext_vector_type(8)))  float    v8f;
typedef __attribute__((ext_vector_type(4)))  unsigned int u32x4;
typedef __attribute__((ext_vector_type(8)))  int      i32x8;
typedef __attribute__((ext_vector_type(4)))  int      i32x4;

constexpr int NB  = 2;
constexpr int NLQ = 2048;
constexpr int NLK = 2048;
constexpr int NH  = 8;
constexpr int ND  = 32;

// ---------------------------------------------------------------------------
// 16-lane (DPP row) butterfly reductions — pure VALU, no LDS traffic.
// quad_perm xor1 (0xB1), quad_perm xor2 (0x4E), row_half_mirror (0x141),
// row_mirror (0x140) together reduce across all 16 lanes of a DPP row.
// ---------------------------------------------------------------------------
__device__ __forceinline__ float row16_max(float x) {
  x = fmaxf(x, __int_as_float(__builtin_amdgcn_update_dpp(0, __float_as_int(x), 0xB1,  0xF, 0xF, true)));
  x = fmaxf(x, __int_as_float(__builtin_amdgcn_update_dpp(0, __float_as_int(x), 0x4E,  0xF, 0xF, true)));
  x = fmaxf(x, __int_as_float(__builtin_amdgcn_update_dpp(0, __float_as_int(x), 0x141, 0xF, 0xF, true)));
  x = fmaxf(x, __int_as_float(__builtin_amdgcn_update_dpp(0, __float_as_int(x), 0x140, 0xF, 0xF, true)));
  return x;
}
__device__ __forceinline__ float row16_sum(float x) {
  x += __int_as_float(__builtin_amdgcn_update_dpp(0, __float_as_int(x), 0xB1,  0xF, 0xF, true));
  x += __int_as_float(__builtin_amdgcn_update_dpp(0, __float_as_int(x), 0x4E,  0xF, 0xF, true));
  x += __int_as_float(__builtin_amdgcn_update_dpp(0, __float_as_int(x), 0x141, 0xF, 0xF, true));
  x += __int_as_float(__builtin_amdgcn_update_dpp(0, __float_as_int(x), 0x140, 0xF, 0xF, true));
  return x;
}

// ---------------------------------------------------------------------------
// TDM: DMA one noise tile (x=32 k-floats, y=16 q-rows, z=2 streams) into LDS.
// D# packed per CDNA5 ISA §8: group0 = count/lds_addr/global_addr/type,
// group1 = data_size + dims/strides/tile dims, group2 = tensor_dim2.
// This toolchain exposes the 6-arg builtin: (g0, g1, g2, g3, g4, cpol).
// ---------------------------------------------------------------------------
__device__ __forceinline__ void tdm_noise_load(const float* gptr, unsigned lds_off) {
  unsigned long long ga = (unsigned long long)(uintptr_t)gptr;
  u32x4 g0 = {
      1u,                                            // count=1 (valid user D#)
      lds_off,                                       // lds_addr (bytes)
      (unsigned)(ga & 0xFFFFFFFFull),                // global_addr[31:0]
      (unsigned)((ga >> 32) & 0x1FFFFFFull) | (2u << 30)  // addr[56:32] | type=2
  };
  i32x8 g1 = {
      (int)(2u << 16),                               // wg_mask=0, data_size=4B
      (int)((unsigned)NLK << 16),                    // tensor_dim0[15:0]=2048
      (int)((unsigned)NLQ << 16),                    // dim0 hi=0 | tensor_dim1 lo=2048
      (int)(32u << 16),                              // dim1 hi=0 | tile_dim0=32
      (int)(16u | (2u << 16)),                       // tile_dim1=16 | tile_dim2=2
      (int)NLK,                                      // tensor_dim0_stride = 2048
      0,                                             // stride0 hi | dim1_stride[15:0]=0
      (int)(((unsigned long long)NLQ * NLK) >> 16)   // tensor_dim1_stride[47:16] (LQ*LK)
  };
  i32x4 g2 = { 2, 0, 0, 0 };                         // tensor_dim2 = 2 streams
  i32x4 g3 = { 0, 0, 0, 0 };
  i32x8 g4 = { 0, 0, 0, 0, 0, 0, 0, 0 };             // unused trailing group
  __builtin_amdgcn_tensor_load_to_lds(g0, g1, g2, g3, g4, 0);
}

// ---------------------------------------------------------------------------
// Projection: Y[r, c] = sum_k X[r*Kin+k] * W[k*32+c], f16 output (row-major)
// ---------------------------------------------------------------------------
__global__ void proj_qk(const float* __restrict__ X, const float* __restrict__ W,
                        _Float16* __restrict__ Y, int R, int Kin) {
  int idx = blockIdx.x * blockDim.x + threadIdx.x;
  if (idx >= R * 32) return;
  int r = idx >> 5, c = idx & 31;
  const float* xr = X + (size_t)r * Kin;
  float acc = 0.0f;
  for (int k = 0; k < Kin; ++k) acc = fmaf(xr[k], W[k * 32 + c], acc);
  Y[(size_t)r * 32 + c] = (_Float16)acc;
}

// V projection, stored transposed: Vt[b][h][d][k] (WMMA B-fragments contiguous).
__global__ void proj_v(const float* __restrict__ X, const float* __restrict__ Wv,
                       _Float16* __restrict__ Vt) {
  int idx = blockIdx.x * blockDim.x + threadIdx.x;
  if (idx >= NB * NLK * 256) return;
  int hd = idx & 255;
  int bk = idx >> 8;
  int b  = bk / NLK, k = bk % NLK;
  const float* xr = X + (size_t)bk * 256;
  float acc = 0.0f;
  for (int kk = 0; kk < 256; ++kk) acc = fmaf(xr[kk], Wv[kk * 256 + hd], acc);
  Vt[(((size_t)b * NH + (hd >> 5)) * ND + (hd & 31)) * NLK + k] = (_Float16)acc;
}

// ---------------------------------------------------------------------------
// Flash attention: block = 8 waves (one head each), 16-query tile per block,
// 32-key tiles. TDM double-buffers the shared noise tile into LDS.
// ---------------------------------------------------------------------------
__global__ __launch_bounds__(256) void attn_kernel(
    const _Float16* __restrict__ Qit, const _Float16* __restrict__ Kit,
    const _Float16* __restrict__ Qc,  const _Float16* __restrict__ Kc,
    const _Float16* __restrict__ Vt,
    const float* __restrict__ noise, const float* __restrict__ sigma,
    const float* __restrict__ Whdp,  const float* __restrict__ bhdp,
    const int* __restrict__ key_mask, const int* __restrict__ query_mask,
    const float* __restrict__ qin, float* __restrict__ out) {
  __shared__ _Float16 ldsP[8][16][32];        // per-wave P-tile transpose (8 KB)
  __shared__ float    nb[2][2][16][32];       // double-buffered noise tile (8 KB)

  const int wave = threadIdx.x >> 5;
  const int lane = threadIdx.x & 31;
  const int col  = lane & 15;                 // A-row / B-col / C-col index
  const int hi   = lane >> 4;                 // lane half (per ISA VGPR layouts)
  const int h    = wave;                      // head
  const int b    = blockIdx.y;
  const int qbase = blockIdx.x * 16;

  const float inv = 0.17677669529663687f;     // 1/sqrt(32)
  const float w0 = Whdp[h], w1 = Whdp[NH + h];
  float sg0 = sigma[b * 2 + 0]; sg0 *= sg0;
  float sg1 = sigma[b * 2 + 1]; sg1 *= sg1;
  const float c_it = w0 * inv, c_ctx = w1 * inv, c_b = bhdp[h] * inv;
  const float c_n0 = w0 * sg0 * inv, c_n1 = w1 * sg1 * inv;

  // A-operand fragments of the Q tile (row = lane&15; K split by lane half).
  v16h aQit, aQc;
  {
    const _Float16* q  = Qit + (size_t)(b * NLQ + qbase + col) * 32;
    const _Float16* qc = Qc  + (size_t)(b * NLQ + qbase + col) * 32;
    v8h q0  = *(const v8h*)(q  + hi * 16);
    v8h q1  = *(const v8h*)(q  + hi * 16 + 8);
    v8h qc0 = *(const v8h*)(qc + hi * 16);
    v8h qc1 = *(const v8h*)(qc + hi * 16 + 8);
#pragma unroll
    for (int i = 0; i < 8; ++i) {
      aQit[i] = q0[i];  aQit[8 + i] = q1[i];
      aQc[i]  = qc0[i]; aQc[8 + i]  = qc1[i];
    }
  }

  // Online-softmax state (C-layout: register j holds row j + 8*hi).
  v8f acc0 = {}, acc1 = {};
  float m[8], l[8];
#pragma unroll
  for (int j = 0; j < 8; ++j) { m[j] = -3.0e38f; l[j] = 0.0f; }

  const float* noise_base = noise + (size_t)(b * 2) * NLQ * NLK + (size_t)qbase * NLK;

  // Preload first noise tile via TDM (wave 0 issues; TENSORcnt tracks it).
  if (wave == 0) {
    tdm_noise_load(noise_base, (unsigned)(uintptr_t)&nb[0][0][0][0]);
  }

  const int NIT = NLK / 32;
  for (int it = 0; it < NIT; ++it) {
    const int kb  = it * 32;
    const int cur = it & 1;

    if (wave == 0) {
      if (it + 1 < NIT) {
        tdm_noise_load(noise_base + kb + 32, (unsigned)(uintptr_t)&nb[cur ^ 1][0][0][0]);
        __builtin_amdgcn_s_wait_tensorcnt(1);   // current tile done, prefetch in flight
      } else {
        __builtin_amdgcn_s_wait_tensorcnt(0);
      }
    }
    __syncthreads();                            // nb[cur] visible to all waves

    // --- B-operand fragments of K tiles: b[i] = Keys[key, d=i+16*hi] -------
    const v16h bKit0 = *(const v16h*)(Kit + (size_t)(b * NLK + kb + col)      * 32 + hi * 16);
    const v16h bKit1 = *(const v16h*)(Kit + (size_t)(b * NLK + kb + 16 + col) * 32 + hi * 16);
    const v16h bKc0  = *(const v16h*)(Kc  + (size_t)(b * NLK + kb + col)      * 32 + hi * 16);
    const v16h bKc1  = *(const v16h*)(Kc  + (size_t)(b * NLK + kb + 16 + col) * 32 + hi * 16);

    v8f z = {};
    v8f sit0 = __builtin_amdgcn_wmma_f32_16x16x32_f16(false, aQit, false, bKit0, (short)0, z, false, false);
    v8f sit1 = __builtin_amdgcn_wmma_f32_16x16x32_f16(false, aQit, false, bKit1, (short)0, z, false, false);
    v8f sct0 = __builtin_amdgcn_wmma_f32_16x16x32_f16(false, aQc,  false, bKc0,  (short)0, z, false, false);
    v8f sct1 = __builtin_amdgcn_wmma_f32_16x16x32_f16(false, aQc,  false, bKc1,  (short)0, z, false, false);

    const int km0 = key_mask[b * NLK + kb + col];
    const int km1 = key_mask[b * NLK + kb + 16 + col];

#pragma unroll
    for (int j = 0; j < 8; ++j) {
      const int row = j + hi * 8;
      // Noise comes from the TDM-staged LDS tile (shared by all 8 heads).
      const float n00 = nb[cur][0][row][col];
      const float n10 = nb[cur][1][row][col];
      const float n01 = nb[cur][0][row][16 + col];
      const float n11 = nb[cur][1][row][16 + col];
      float sc0 = sit0[j] * c_it + sct0[j] * c_ctx + c_b + n00 * c_n0 + n10 * c_n1;
      float sc1 = sit1[j] * c_it + sct1[j] * c_ctx + c_b + n01 * c_n0 + n11 * c_n1;
      if (!km0) sc0 = -1.0e30f;
      if (!km1) sc1 = -1.0e30f;

      // Row reduction within the 16-lane half via DPP butterflies.
      const float mnew  = fmaxf(m[j], row16_max(fmaxf(sc0, sc1)));
      const float alpha = __expf(m[j] - mnew);
      const float p0 = __expf(sc0 - mnew);
      const float p1 = __expf(sc1 - mnew);
      const float rs = row16_sum(p0 + p1);
      l[j] = l[j] * alpha + rs;
      m[j] = mnew;
      acc0[j] = acc0[j] * alpha;
      acc1[j] = acc1[j] * alpha;

      // C-layout -> LDS so we can refetch P in the A-operand layout.
      ldsP[wave][row][col]      = (_Float16)p0;
      ldsP[wave][row][16 + col] = (_Float16)p1;
    }

    // Same-wave LDS store->load ordering (DS ops are in-order; force the wait).
    asm volatile("s_wait_dscnt 0" ::: "memory");

    v16h aP;
    {
      v8h p0 = *(const v8h*)&ldsP[wave][col][hi * 16];
      v8h p1 = *(const v8h*)&ldsP[wave][col][hi * 16 + 8];
#pragma unroll
      for (int i = 0; i < 8; ++i) { aP[i] = p0[i]; aP[8 + i] = p1[i]; }
    }

    // B-operand fragments of V: b[i] = Vt[b][h][d = noff+col][kb + i + 16*hi]
    const _Float16* vb = Vt + ((size_t)b * NH + h) * ND * NLK + kb + hi * 16;
    const v16h bV0 = *(const v16h*)(vb + (size_t)col * NLK);
    const v16h bV1 = *(const v16h*)(vb + (size_t)(16 + col) * NLK);

    acc0 = __builtin_amdgcn_wmma_f32_16x16x32_f16(false, aP, false, bV0, (short)0, acc0, false, false);
    acc1 = __builtin_amdgcn_wmma_f32_16x16x32_f16(false, aP, false, bV1, (short)0, acc1, false, false);

    __syncthreads();                            // all waves done with nb[cur]
  }

  // Epilogue: out = qin + (acc / l) * query_mask    (layout [b, q, h*32 + d])
#pragma unroll
  for (int j = 0; j < 8; ++j) {
    const int q = qbase + j + hi * 8;
    const float qm = (float)query_mask[b * NLQ + q];
    const float s = (l[j] > 0.0f) ? (qm / l[j]) : 0.0f;
    const size_t o = (size_t)(b * NLQ + q) * 256 + h * 32;
    out[o + col]      = qin[o + col]      + acc0[j] * s;
    out[o + 16 + col] = qin[o + 16 + col] + acc1[j] * s;
  }
}

// ---------------------------------------------------------------------------
extern "C" void kernel_launch(void* const* d_in, const int* in_sizes, int n_in,
                              void* d_out, int out_size, void* d_ws, size_t ws_size,
                              hipStream_t stream) {
  const float* queries_it  = (const float*)d_in[0];
  const float* queries_ctx = (const float*)d_in[1];
  const float* keys_it     = (const float*)d_in[2];
  const float* keys_ctx    = (const float*)d_in[3];
  const float* sigma_noise = (const float*)d_in[4];
  const float* noise       = (const float*)d_in[5];
  const float* Wq_it       = (const float*)d_in[6];
  const float* Wk_it       = (const float*)d_in[7];
  const float* Wq_ctx      = (const float*)d_in[8];
  const float* Wk_ctx      = (const float*)d_in[9];
  const float* Wv          = (const float*)d_in[10];
  const float* W_hdp       = (const float*)d_in[11];
  const float* b_hdp       = (const float*)d_in[12];
  const int*   key_mask    = (const int*)d_in[13];
  const int*   query_mask  = (const int*)d_in[14];

  // Workspace: f16 Qit/Kit/Qc/Kc [B*L,32] + Vt [B,H,D,LK]  (~3 MB total)
  _Float16* Qit = (_Float16*)d_ws;
  _Float16* Kit = Qit + (size_t)NB * NLQ * 32;
  _Float16* Qc  = Kit + (size_t)NB * NLK * 32;
  _Float16* Kc  = Qc  + (size_t)NB * NLQ * 32;
  _Float16* Vt  = Kc  + (size_t)NB * NLK * 32;

  const int R = NB * NLQ;
  const int tpb = 256;
  proj_qk<<<(R * 32 + tpb - 1) / tpb, tpb, 0, stream>>>(queries_it,  Wq_it,  Qit, R, 256);
  proj_qk<<<(R * 32 + tpb - 1) / tpb, tpb, 0, stream>>>(keys_it,     Wk_it,  Kit, R, 256);
  proj_qk<<<(R * 32 + tpb - 1) / tpb, tpb, 0, stream>>>(queries_ctx, Wq_ctx, Qc,  R, 128);
  proj_qk<<<(R * 32 + tpb - 1) / tpb, tpb, 0, stream>>>(keys_ctx,    Wk_ctx, Kc,  R, 128);
  proj_v<<<(NB * NLK * 256 + tpb - 1) / tpb, tpb, 0, stream>>>(keys_it, Wv, Vt);

  dim3 grid(NLQ / 16, NB);
  attn_kernel<<<grid, 256, 0, stream>>>(Qit, Kit, Qc, Kc, Vt,
                                        noise, sigma_noise, W_hdp, b_hdp,
                                        key_mask, query_mask,
                                        queries_it, (float*)d_out);
}
